// GCN_25555055411697
// MI455X (gfx1250) — matrix operand order repaired and checked by reading, not compiled, not run on hardware
//
#include <hip/hip_runtime.h>
#include <hip/hip_bf16.h>

typedef _Float16 half16 __attribute__((ext_vector_type(16)));
typedef float    float8 __attribute__((ext_vector_type(8)));

#define N_NODES 100000
#define N_EDGES 1600000
#define C_IN    128
#define C_HID   128
#define C_OUT   64
#define BN_EPS  1e-5f

// ---------------------------------------------------------------------------
// WMMA fragment-layout helpers (CDNA5 16-bit layouts, ISA §7.12.2).
// A-matrix 16x32 f16: lane = hi*16 + m (m = row%16).
//   halves 0..7  <-> K = k0 + hi*8 + i        (i=0..7)
//   halves 8..15 <-> K = k0 + hi*8 + 16 + i-8
// B-matrix 32x16 f16: lane = hi*16 + n (n = col%16), same K mapping as A-matrix.
// Swizzled storage: frag block = [tile][kgroup][lane][16 halves] (32B per lane).
// ---------------------------------------------------------------------------
__device__ __forceinline__ size_t a_frag_index(int row, int k, int K) {
    int rt = row >> 4, m = row & 15;
    int kg = k >> 5, ko = k & 31;
    int hi, i;
    if (ko < 8)       { hi = 0; i = ko; }
    else if (ko < 16) { hi = 1; i = ko - 8; }
    else if (ko < 24) { hi = 0; i = ko - 8; }
    else              { hi = 1; i = ko - 16; }
    int lane = hi * 16 + m;
    return (((size_t)rt * (K >> 5) + kg) * 32 + lane) * 16 + i;
}

__device__ __forceinline__ size_t b_frag_index(int k, int n, int K) {
    int nt = n >> 4, c = n & 15;
    int kg = k >> 5, ko = k & 31;
    int hi = (ko >> 3) & 1, i = (ko & 7) | ((ko >> 4) << 3);
    int lane = hi * 16 + c;
    return (((size_t)nt * (K >> 5) + kg) * 32 + lane) * 16 + i;
}

// ---------------- pack A (row-major f32 [rows,K]) -> swizzled f16 fragments ----------------
__global__ void k_pack_a(const float* __restrict__ in, _Float16* __restrict__ out,
                         int rows, int K) {
    size_t total = (size_t)rows * K;
    size_t idx = (size_t)blockIdx.x * blockDim.x + threadIdx.x;
    size_t stride = (size_t)gridDim.x * blockDim.x;
    for (; idx < total; idx += stride) {
        int row = (int)(idx / K);
        int k   = (int)(idx - (size_t)row * K);
        out[a_frag_index(row, k, K)] = (_Float16)in[idx];
    }
}

// ---------------- pack B (row-major f32 [K,NB]) -> swizzled f16 fragments ----------------
__global__ void k_pack_b(const float* __restrict__ in, _Float16* __restrict__ out,
                         int K, int NB) {
    size_t total = (size_t)K * NB;
    size_t idx = (size_t)blockIdx.x * blockDim.x + threadIdx.x;
    size_t stride = (size_t)gridDim.x * blockDim.x;
    for (; idx < total; idx += stride) {
        int k = (int)(idx / NB);
        int n = (int)(idx - (size_t)k * NB);
        out[b_frag_index(k, n, K)] = (_Float16)in[idx];
    }
}

// ---------------- weighted in-degree via edge atomics ----------------
__global__ void k_deg_edges(const int* __restrict__ ei, const float* __restrict__ w,
                            float* __restrict__ deg, int E) {
    int e = blockIdx.x * blockDim.x + threadIdx.x;
    if (e < E) atomicAdd(&deg[ei[E + e]], w[e]);   // col = target
}

__global__ void k_finalize_dinv(float* __restrict__ deg, int n) {
    int i = blockIdx.x * blockDim.x + threadIdx.x;
    if (i < n) {
        float d = deg[i] + 1.0f;                   // + self-loop weight 1
        deg[i] = (d > 0.0f) ? rsqrtf(d) : 0.0f;
    }
}

// ---------------- WMMA GEMM on pre-swizzled fragments ----------------
// C[MxNB] = A[MxK] * B[KxNB]; blockDim=128 (4 waves), each wave one 16x16 tile.
template <int K, int NB>
__global__ __launch_bounds__(128) void k_gemm_wmma(const _Float16* __restrict__ As,
                                                   const _Float16* __restrict__ Bs,
                                                   float* __restrict__ C) {
    const int lane = threadIdx.x & 31;
    const int wave = threadIdx.x >> 5;
    const int m    = lane & 15;
    const int hi   = lane >> 4;
    const int rt   = blockIdx.x;                   // row tile
    const int nt   = blockIdx.y * 4 + wave;        // col tile
    constexpr int KG = K >> 5;

    const _Float16* ap = As + (((size_t)rt * KG) * 32 + lane) * 16;
    const _Float16* bp = Bs + (((size_t)nt * KG) * 32 + lane) * 16;

    float8 acc = {};
#pragma unroll
    for (int kg = 0; kg < KG; ++kg) {
        half16 a = *(const half16*)(ap + (size_t)kg * 32 * 16);
        half16 b = *(const half16*)(bp + (size_t)kg * 32 * 16);
        acc = __builtin_amdgcn_wmma_f32_16x16x32_f16(
            /*neg_a=*/false, a, /*neg_b=*/false, b,
            /*c_mod=*/(short)0, acc, /*reuse_a=*/false, /*reuse_b=*/false);
    }
    // D layout: VGPR r -> row (hi*8 + r), col nt*16 + m
    const int rowBase = rt * 16 + hi * 8;
    const int n = nt * 16 + m;
#pragma unroll
    for (int r = 0; r < 8; ++r)
        C[(size_t)(rowBase + r) * NB + n] = acc[r];
}

// ---------------- out[n,c] = dinv[n]^2 * h[n,c] + bias[c]  (self-loop + bias init) ----------------
__global__ void k_self_init(const float* __restrict__ h, const float* __restrict__ dinv,
                            const float* __restrict__ bias, float* __restrict__ out,
                            int C, size_t total) {
    size_t i = (size_t)blockIdx.x * blockDim.x + threadIdx.x;
    size_t stride = (size_t)gridDim.x * blockDim.x;
    for (; i < total; i += stride) {
        size_t n = i / C;
        int    c = (int)(i - n * C);
        float di = dinv[n];
        out[i] = di * di * h[i] + bias[c];
    }
}

// ---------------- edge aggregation: out[col] += dinv[row]*w*dinv[col] * h[row] ----------------
__global__ void k_agg_edges(const float* __restrict__ h, const int* __restrict__ ei,
                            const float* __restrict__ w, const float* __restrict__ dinv,
                            float* __restrict__ out, int C, int E) {
    int e = blockIdx.x * (blockDim.x >> 5) + (threadIdx.x >> 5);
    if (e >= E) return;
    int lane = threadIdx.x & 31;
    int row = ei[e];
    int col = ei[E + e];
    float norm = dinv[row] * w[e] * dinv[col];
    const float* hp = h + (size_t)row * C;
    float* op = out + (size_t)col * C;
    for (int c = lane; c < C; c += 32)
        atomicAdd(&op[c], hp[c] * norm);
}

// ---------------- BatchNorm statistics: per-channel sum & sumsq ----------------
__global__ void k_bn_stats(const float* __restrict__ h, float* __restrict__ stats,
                           int n, int rowsPerBlock) {
    int c = threadIdx.x;                 // 128 threads = 128 channels
    int r0 = blockIdx.x * rowsPerBlock;
    int r1 = r0 + rowsPerBlock; if (r1 > n) r1 = n;
    float s = 0.0f, q = 0.0f;
    for (int r = r0; r < r1; ++r) {
        float v = h[(size_t)r * C_HID + c];
        s += v;
        q += v * v;
    }
    atomicAdd(&stats[c], s);
    atomicAdd(&stats[C_HID + c], q);
}

// ---------------- BN apply + ReLU, emitted directly in swizzled f16 fragment layout ----------------
__global__ void k_bn_relu_pack(const float* __restrict__ h, const float* __restrict__ stats,
                               const float* __restrict__ gamma, const float* __restrict__ beta,
                               _Float16* __restrict__ out, size_t total) {
    size_t i = (size_t)blockIdx.x * blockDim.x + threadIdx.x;
    size_t stride = (size_t)gridDim.x * blockDim.x;
    const float invN = 1.0f / (float)N_NODES;
    for (; i < total; i += stride) {
        int c = (int)(i & (C_HID - 1));        // channel == K index of next GEMM
        int row = (int)(i >> 7);
        float mean = stats[c] * invN;
        float var  = stats[C_HID + c] * invN - mean * mean;
        float v = (h[i] - mean) * rsqrtf(var + BN_EPS) * gamma[c] + beta[c];
        out[a_frag_index(row, c, C_HID)] = (_Float16)fmaxf(v, 0.0f);
    }
}

extern "C" void kernel_launch(void* const* d_in, const int* in_sizes, int n_in,
                              void* d_out, int out_size, void* d_ws, size_t ws_size,
                              hipStream_t stream) {
    const float* x     = (const float*)d_in[0];
    const int*   ei    = (const int*)  d_in[1];   // [2, E] row-major: rows, then cols
    const float* ew    = (const float*)d_in[2];
    const float* W1    = (const float*)d_in[3];
    const float* b1    = (const float*)d_in[4];
    const float* gamma = (const float*)d_in[5];
    const float* beta  = (const float*)d_in[6];
    const float* W2    = (const float*)d_in[7];
    const float* b2    = (const float*)d_in[8];
    float* out = (float*)d_out;

    // ---- carve workspace (256B aligned) ----
    char* p = (char*)d_ws;
    auto carve = [&](size_t bytes) { char* r = p; p += (bytes + 255) & ~(size_t)255; return (void*)r; };
    float*     dinv  = (float*)    carve((size_t)N_NODES * 4);
    _Float16*  Xs    = (_Float16*) carve((size_t)N_NODES * C_IN * 2);   // swizzled A fragments
    _Float16*  W1s   = (_Float16*) carve((size_t)C_IN * C_HID * 2);     // swizzled B fragments
    _Float16*  W2s   = (_Float16*) carve((size_t)C_HID * C_OUT * 2);
    float*     H     = (float*)    carve((size_t)N_NODES * C_HID * 4);  // GEMM1 out; reused for GEMM2 out
    float*     A1    = (float*)    carve((size_t)N_NODES * C_HID * 4);  // aggregated layer-1 features
    float*     stats = (float*)    carve(2 * C_HID * 4);
    _Float16*  Hact  = Xs;                                              // reuse after GEMM1 consumed Xs

    hipMemsetAsync(dinv,  0, (size_t)N_NODES * 4, stream);
    hipMemsetAsync(stats, 0, 2 * C_HID * 4, stream);

    // ---- pack operands to swizzled f16 fragments ----
    k_pack_a<<<4096, 256, 0, stream>>>(x,  Xs,  N_NODES, C_IN);
    k_pack_b<<<64,   256, 0, stream>>>(W1, W1s, C_IN,  C_HID);
    k_pack_b<<<32,   256, 0, stream>>>(W2, W2s, C_HID, C_OUT);

    // ---- degree / normalization ----
    k_deg_edges<<<(N_EDGES + 255) / 256, 256, 0, stream>>>(ei, ew, dinv, N_EDGES);
    k_finalize_dinv<<<(N_NODES + 255) / 256, 256, 0, stream>>>(dinv, N_NODES);

    // ---- layer 1: linear (WMMA) ----
    dim3 g1(N_NODES / 16, C_HID / 64);
    k_gemm_wmma<C_IN, C_HID><<<g1, 128, 0, stream>>>(Xs, W1s, H);

    // ---- layer 1: aggregation (self-loop+bias init, then edges) ----
    k_self_init<<<4096, 256, 0, stream>>>(H, dinv, b1, A1, C_HID, (size_t)N_NODES * C_HID);
    k_agg_edges<<<(N_EDGES + 7) / 8, 256, 0, stream>>>(H, ei, ew, dinv, A1, C_HID, N_EDGES);

    // ---- BatchNorm + ReLU -> swizzled f16 activations ----
    const int rowsPerBlock = 500;  // 200 blocks * 500 rows = 100000
    k_bn_stats<<<(N_NODES + rowsPerBlock - 1) / rowsPerBlock, C_HID, 0, stream>>>(A1, stats, N_NODES, rowsPerBlock);
    k_bn_relu_pack<<<4096, 256, 0, stream>>>(A1, stats, gamma, beta, Hact, (size_t)N_NODES * C_HID);

    // ---- layer 2: linear (WMMA, 128 -> 64) ----
    dim3 g2(N_NODES / 16, C_OUT / 64);
    k_gemm_wmma<C_HID, C_OUT><<<g2, 128, 0, stream>>>(Hact, W2s, H);

    // ---- layer 2: aggregation straight into d_out ----
    k_self_init<<<4096, 256, 0, stream>>>(H, dinv, b2, out, C_OUT, (size_t)N_NODES * C_OUT);
    k_agg_edges<<<(N_EDGES + 7) / 8, 256, 0, stream>>>(H, ei, ew, dinv, out, C_OUT, N_EDGES);
}
